// CrossAttention_24094766530836
// MI455X (gfx1250) — compile-verified, matrix operand order checked by reading, not run on hardware
//
#include <hip/hip_runtime.h>
#include <hip/hip_bf16.h>

// ---------------------------------------------------------------------------
// CDNA5 (gfx1250) implementation: all matmuls via v_wmma_f32_16x16x32_bf16
// (wave32, f32 accumulate); attention Q/K tiles staged by the Tensor Data
// Mover (tensor_load_to_lds + s_wait_tensorcnt).
// ---------------------------------------------------------------------------

typedef __bf16 bf16;
typedef __attribute__((ext_vector_type(8)))  __bf16 bf16x8;
typedef __attribute__((ext_vector_type(16))) __bf16 bf16x16;
typedef __attribute__((ext_vector_type(8)))  float  f32x8;
typedef __attribute__((ext_vector_type(4)))  float  f32x4;
typedef __attribute__((ext_vector_type(4)))  unsigned int u32x4;
typedef __attribute__((ext_vector_type(8)))  unsigned int u32x8;

// Load one 16x32 bf16 WMMA operand fragment from an LDS image laid out
// [row][k] (A: row=M, "B": row=N i.e. B stored transposed). Per ISA 7.12.2:
// lane L (0..15) holds row L; lanes 16..31 hold the +8 K offset; elements
// 0..7 cover K=kbase+half*8.., elements 8..15 cover K=kbase+16+half*8..
__device__ __forceinline__ bf16x16 frag_ld(const bf16* base, int strideElems,
                                           int row, int kbase, int lane) {
  const bf16* p = base + (size_t)row * strideElems + kbase + ((lane >> 4) << 3);
  bf16x8 lo = *(const bf16x8*)(p);
  bf16x8 hi = *(const bf16x8*)(p + 16);
  return __builtin_shufflevector(lo, hi, 0,1,2,3,4,5,6,7,8,9,10,11,12,13,14,15);
}

__device__ __forceinline__ f32x8 wmma_bf16f32(bf16x16 a, bf16x16 b, f32x8 c) {
  return __builtin_amdgcn_wmma_f32_16x16x32_bf16(false, a, false, b,
                                                 (short)0, c, false, false);
}

// Tensor Data Mover: async DMA a [rows x 64] bf16 tile (row pitch `pitchElems`
// in global) into LDS at byte offset lds_off, padded to a 72-element (144 B)
// LDS row stride via the TDM pad feature (pad_interval=32 DW, pad_amount=4 DW).
// Issued once per wave (EXEC of the calling wave must be non-zero).
__device__ __forceinline__ void tdm_load_tile64(unsigned lds_off,
                                                const void* gptr,
                                                int rows, int pitchElems) {
  const unsigned long long ga = (unsigned long long)gptr;
  u32x4 g0;
  g0[0] = 1u;                                        // count = 1 (valid D#)
  g0[1] = lds_off;                                   // LDS dest byte offset
  g0[2] = (unsigned)(ga & 0xFFFFFFFFu);              // global_addr[31:0]
  g0[3] = (unsigned)((ga >> 32) & 0x01FFFFFFu)       // global_addr[56:32]
        | (2u << 30);                                // type = 2 ("image")
  u32x8 g1;
  g1[0] = (1u << 16)                                 // data_size = 1 -> 2 bytes
        | (1u << 20)                                 // pad_enable
        | (4u << 22)                                 // pad_interval: 32 DWORDs
        | (3u << 25);                                // pad_amount: 4 DWORDs
  g1[1] = 64u << 16;                                 // tensor_dim0 = 64
  g1[2] = ((unsigned)rows) << 16;                    // tensor_dim1 = rows
  g1[3] = 64u << 16;                                 // tile_dim0 = 64
  g1[4] = (unsigned)rows;                            // tile_dim1 = rows
  g1[5] = (unsigned)pitchElems;                      // tensor_dim0_stride
  g1[6] = 0u;
  g1[7] = 0u;
  asm volatile("tensor_load_to_lds %0, %1" :: "s"(g0), "s"(g1) : "memory");
}

// ---------------------------------------------------------------------------
// Generic 128x128x32-tiled GEMM, 256 threads (8 waves), wave tile 32x64.
// AMODE 0: A is row-major f32 [M][K].
// AMODE 1: A rows are im2col-gathered 2x2/stride-2 patches of v [4][3136][512],
//          K = 2048, kk = (kh*2+kw)*512 + ci.
// BMODE 0: B is row-major f32 [K][N] (weights stored [in,out]).
// BMODE 1: B[kk][co] = sr_w OIHW: sr_w[co*2048 + ci*4 + (kh*2+kw)].
// OUT_BF16: write bf16 instead of f32.  ADD_BIAS: += bias[n].
// ---------------------------------------------------------------------------
template <int AMODE, int BMODE, int OUT_BF16, int ADD_BIAS>
__global__ void __launch_bounds__(256)
gemm_wmma_kernel(const float* __restrict__ A, const float* __restrict__ B,
                 const float* __restrict__ bias, void* __restrict__ Cout,
                 int M, int N, int K) {
  __shared__ __align__(16) bf16 Alds[128][40];   // [m][k], 80B row stride
  __shared__ __align__(16) bf16 Blds[128][40];   // [n][k] (B transposed)

  const int t = threadIdx.x;
  const int lane = t & 31;
  const int wave = t >> 5;
  const int waveM = wave >> 1;             // 0..3 -> 32 rows each
  const int waveN = wave & 1;              // 0..1 -> 64 cols each
  const int m0 = blockIdx.x * 128;
  const int n0 = blockIdx.y * 128;

  f32x8 acc[2][4];
#pragma unroll
  for (int mt = 0; mt < 2; ++mt)
#pragma unroll
    for (int nt = 0; nt < 4; ++nt)
#pragma unroll
      for (int r = 0; r < 8; ++r) acc[mt][nt][r] = 0.f;

  const int arow = t >> 1;                 // 0..127
  const int acol = (t & 1) << 4;           // 0 / 16
  const int kl   = t & 31;                 // B loader: k within tile
  const int nb   = wave << 4;              // B loader: n-group base (0..112)

  for (int k0 = 0; k0 < K; k0 += 32) {
    // ---- stage A tile -> LDS (f32 -> bf16) ----
    float av[16];
    const int gRow = m0 + arow;
    if (gRow < M) {
      const float* ap;
      if (AMODE == 0) {
        ap = A + (size_t)gRow * K + k0 + acol;
      } else {
        const int bidx = gRow / 784;
        const int tkv  = gRow - bidx * 784;
        const int oh = tkv / 28, ow = tkv - (tkv / 28) * 28;
        const int kk = k0 + acol;          // 16 consecutive kk share one patch
        const int p  = kk >> 9;
        const int ci = kk & 511;
        const int token = ((oh << 1) + (p >> 1)) * 56 + ((ow << 1) + (p & 1));
        ap = A + ((size_t)bidx * 3136 + token) * 512 + ci;
      }
#pragma unroll
      for (int i = 0; i < 4; ++i) {
        f32x4 q4 = *(const f32x4*)(ap + 4 * i);
#pragma unroll
        for (int j = 0; j < 4; ++j) av[4 * i + j] = q4[j];
      }
      if (k0 + 32 < K) __builtin_prefetch(ap + 32, 0, 1);  // global_prefetch
    } else {
#pragma unroll
      for (int i = 0; i < 16; ++i) av[i] = 0.f;
    }
    {
      bf16x8 lo, hi;
#pragma unroll
      for (int e = 0; e < 8; ++e) { lo[e] = (bf16)av[e]; hi[e] = (bf16)av[e + 8]; }
      *(bf16x8*)&Alds[arow][acol]     = lo;
      *(bf16x8*)&Alds[arow][acol + 8] = hi;
    }

    // ---- stage B tile -> LDS transposed [n][k] ----
    if (BMODE == 0) {
      const float* bp = B + (size_t)(k0 + kl) * N + n0 + nb;
#pragma unroll
      for (int i = 0; i < 4; ++i) {
        f32x4 q4 = *(const f32x4*)(bp + 4 * i);
#pragma unroll
        for (int j = 0; j < 4; ++j) Blds[nb + 4 * i + j][kl] = (bf16)q4[j];
      }
    } else {
      const int kk = k0 + kl;
      const int p  = kk >> 9;
      const int ci = kk & 511;
#pragma unroll
      for (int i = 0; i < 16; ++i) {
        const int co = n0 + nb + i;
        Blds[nb + i][kl] = (bf16)B[(size_t)co * 2048 + (ci << 2) + p];
      }
    }
    __syncthreads();

    // ---- WMMA: 2 (M) x 4 (N) fragments per wave ----
    bf16x16 afr[2], bfr[4];
#pragma unroll
    for (int mt = 0; mt < 2; ++mt)
      afr[mt] = frag_ld(&Alds[0][0], 40, waveM * 32 + mt * 16 + (lane & 15), 0, lane);
#pragma unroll
    for (int nt = 0; nt < 4; ++nt)
      bfr[nt] = frag_ld(&Blds[0][0], 40, waveN * 64 + nt * 16 + (lane & 15), 0, lane);
#pragma unroll
    for (int mt = 0; mt < 2; ++mt)
#pragma unroll
      for (int nt = 0; nt < 4; ++nt)
        acc[mt][nt] = wmma_bf16f32(afr[mt], bfr[nt], acc[mt][nt]);
    __syncthreads();
  }

  // ---- epilogue: C layout -> lane n = lane&15, row = r + 8*(lane>=16) ----
#pragma unroll
  for (int mt = 0; mt < 2; ++mt)
#pragma unroll
    for (int nt = 0; nt < 4; ++nt) {
      const int gn = n0 + waveN * 64 + nt * 16 + (lane & 15);
      const float bv = ADD_BIAS ? bias[gn] : 0.f;
#pragma unroll
      for (int r = 0; r < 8; ++r) {
        const int gm = m0 + waveM * 32 + mt * 16 + r + ((lane >> 4) << 3);
        if (gm < M) {
          const float vv = acc[mt][nt][r] + bv;
          if (OUT_BF16) ((bf16*)Cout)[(size_t)gm * N + gn] = (bf16)vv;
          else          ((float*)Cout)[(size_t)gm * N + gn] = vv;
        }
      }
    }
}

// ---------------------------------------------------------------------------
// LayerNorm over c=512 with conv-bias folded in. One 256-thread block per row.
// ---------------------------------------------------------------------------
__global__ void __launch_bounds__(256)
ln_kernel(const float* __restrict__ in, const float* __restrict__ cb,
          const float* __restrict__ g, const float* __restrict__ be,
          float* __restrict__ out) {
  const int row = blockIdx.x;
  const int t = threadIdx.x;
  __shared__ float red[256];
  const float a = in[(size_t)row * 512 + t]       + cb[t];
  const float b = in[(size_t)row * 512 + 256 + t] + cb[256 + t];
  red[t] = a + b;
  __syncthreads();
  for (int s = 128; s > 0; s >>= 1) { if (t < s) red[t] += red[t + s]; __syncthreads(); }
  const float mean = red[0] * (1.f / 512.f);
  __syncthreads();
  const float da = a - mean, db = b - mean;
  red[t] = da * da + db * db;
  __syncthreads();
  for (int s = 128; s > 0; s >>= 1) { if (t < s) red[t] += red[t + s]; __syncthreads(); }
  const float inv = rsqrtf(red[0] * (1.f / 512.f) + 1e-5f);
  out[(size_t)row * 512 + t]       = da * inv * g[t]       + be[t];
  out[(size_t)row * 512 + 256 + t] = db * inv * g[256 + t] + be[256 + t];
}

// ---------------------------------------------------------------------------
// Flash-attention per (b,h): Q[3136,64] x K[784,64]^T -> softmax -> xV[784,64].
// 128 threads (4 waves), 64 q rows per block, kv tiles of 64 (13 tiles).
// Q and K tiles are DMA'd into LDS by the Tensor Data Mover; V goes through
// the vector path because its LDS image must be transposed ([d][kv]).
// ---------------------------------------------------------------------------
__global__ void __launch_bounds__(128)
attention_kernel(const bf16* __restrict__ Q, const bf16* __restrict__ KV,
                 float* __restrict__ O) {
  const int qblk = blockIdx.x;           // 49
  const int bh   = blockIdx.y;           // 32
  const int b    = bh >> 3;
  const int h    = bh & 7;
  const int qbase = qblk * 64;
  const int t = threadIdx.x, lane = t & 31, wave = t >> 5;

  __shared__ __align__(16) bf16  Qs[64][72];     // [q][d]   (TDM-filled)
  __shared__ __align__(16) bf16  Ks[64][72];     // [kv][d]  (TDM-filled)
  __shared__ __align__(16) bf16  Vs[64][72];     // [d][kv]  (B for P.V)
  __shared__ __align__(16) float Sf[64][72];     // scores; P overlays as bf16
  __shared__ float mrow[64], lrow[64], frow[64];
  bf16* Ps = (bf16*)&Sf[0][0];                   // stride 144 bf16 per row

  // Q tile: single TDM 2D load (64 x 64 bf16, pitch 512), issued by wave 0.
  if (wave == 0) {
    tdm_load_tile64((unsigned)(unsigned long long)&Qs[0][0],
                    Q + ((size_t)b * 3136 + qbase) * 512 + h * 64,
                    64, 512);
  }
  if (t < 64) { mrow[t] = -1e30f; lrow[t] = 0.f; }

  f32x8 oacc[4];
#pragma unroll
  for (int nt = 0; nt < 4; ++nt)
#pragma unroll
    for (int r = 0; r < 8; ++r) oacc[nt][r] = 0.f;

  for (int kt = 0; kt < 13; ++kt) {
    const int kvbase = kt * 64;
    const int valid  = (784 - kvbase < 64) ? (784 - kvbase) : 64;
    __syncthreads();                     // previous tile fully consumed

    // K tile: TDM 2D load (valid x 64 bf16, pitch 1024). Rows beyond `valid`
    // keep stale data -> masked to P=0 in the softmax below.
    if (wave == 0) {
      tdm_load_tile64((unsigned)(unsigned long long)&Ks[0][0],
                      KV + ((size_t)b * 784 + kvbase) * 1024 + h * 64,
                      valid, 1024);
    }

    { // V tile, transposed into Vs[d][kv] on the vector path
      const int r = t & 63;              // kv row within tile
      const int c = (t >> 6) << 5;       // d half: 0 / 32
      const int kvtok = kvbase + r;
      if (kvtok < 784) {
        const bf16* vp = KV + ((size_t)b * 784 + kvtok) * 1024 + 512 + h * 64 + c;
#pragma unroll
        for (int i = 0; i < 4; ++i) {
          bf16x8 vv = *(const bf16x8*)(vp + 8 * i);
#pragma unroll
          for (int j = 0; j < 8; ++j) Vs[c + 8 * i + j][r] = vv[j];
        }
      } else {
#pragma unroll
        for (int i = 0; i < 4; ++i)
#pragma unroll
          for (int j = 0; j < 8; ++j) Vs[c + 8 * i + j][r] = (bf16)0.f;
      }
    }
    if (wave == 0) __builtin_amdgcn_s_wait_tensorcnt(0);
    __syncthreads();

    // S = 0.125 * Q K^T  (wave owns 16 q rows x 64 kv)
    bf16x16 qa0 = frag_ld(&Qs[0][0], 72, wave * 16 + (lane & 15), 0, lane);
    bf16x16 qa1 = frag_ld(&Qs[0][0], 72, wave * 16 + (lane & 15), 32, lane);
#pragma unroll
    for (int nt = 0; nt < 4; ++nt) {
      f32x8 s;
#pragma unroll
      for (int r = 0; r < 8; ++r) s[r] = 0.f;
      bf16x16 b0 = frag_ld(&Ks[0][0], 72, nt * 16 + (lane & 15), 0, lane);
      bf16x16 b1 = frag_ld(&Ks[0][0], 72, nt * 16 + (lane & 15), 32, lane);
      s = wmma_bf16f32(qa0, b0, s);
      s = wmma_bf16f32(qa1, b1, s);
      const int cc = nt * 16 + (lane & 15);
#pragma unroll
      for (int r = 0; r < 8; ++r)
        Sf[wave * 16 + r + ((lane >> 4) << 3)][cc] = s[r] * 0.125f;
    }
    __syncthreads();

    // online softmax: one thread per q row (read f32, overlay bf16 P in place)
    if (t < 64) {
      const float mo = mrow[t];
      float mx = mo;
      for (int c = 0; c < valid; ++c) mx = fmaxf(mx, Sf[t][c]);
      const float f = __expf(mo - mx);
      float sum = 0.f;
      for (int c = 0; c < 64; ++c) {
        const float p = (c < valid) ? __expf(Sf[t][c] - mx) : 0.f;
        Ps[t * 144 + c] = (bf16)p;     // safe: 2c <= 4c, element read first
        sum += p;
      }
      mrow[t] = mx;
      lrow[t] = lrow[t] * f + sum;
      frow[t] = f;
    }
    __syncthreads();

    // rescale running accumulators, then O += P V
    float fr[8];
#pragma unroll
    for (int r = 0; r < 8; ++r) fr[r] = frow[wave * 16 + r + ((lane >> 4) << 3)];
#pragma unroll
    for (int nt = 0; nt < 4; ++nt)
#pragma unroll
      for (int r = 0; r < 8; ++r) oacc[nt][r] *= fr[r];

    bf16x16 pa0 = frag_ld(Ps, 144, wave * 16 + (lane & 15), 0, lane);
    bf16x16 pa1 = frag_ld(Ps, 144, wave * 16 + (lane & 15), 32, lane);
#pragma unroll
    for (int nt = 0; nt < 4; ++nt) {
      bf16x16 v0 = frag_ld(&Vs[0][0], 72, nt * 16 + (lane & 15), 0, lane);
      bf16x16 v1 = frag_ld(&Vs[0][0], 72, nt * 16 + (lane & 15), 32, lane);
      oacc[nt] = wmma_bf16f32(pa0, v0, oacc[nt]);
      oacc[nt] = wmma_bf16f32(pa1, v1, oacc[nt]);
    }
  }

  // normalize and write O in [b, n, h*64+d] layout (ready for out-proj GEMM)
  float linv[8];
#pragma unroll
  for (int r = 0; r < 8; ++r)
    linv[r] = 1.f / lrow[wave * 16 + r + ((lane >> 4) << 3)];
#pragma unroll
  for (int nt = 0; nt < 4; ++nt)
#pragma unroll
    for (int r = 0; r < 8; ++r) {
      const int qr = qbase + wave * 16 + r + ((lane >> 4) << 3);
      O[((size_t)b * 3136 + qr) * 512 + h * 64 + nt * 16 + (lane & 15)] =
          oacc[nt][r] * linv[r];
    }
}

// ---------------------------------------------------------------------------
extern "C" void kernel_launch(void* const* d_in, const int* in_sizes, int n_in,
                              void* d_out, int out_size, void* d_ws, size_t ws_size,
                              hipStream_t stream) {
  (void)in_sizes; (void)n_in; (void)out_size; (void)ws_size;
  const float* x      = (const float*)d_in[0];   // [4,3136,512]
  const float* v      = (const float*)d_in[1];   // [4,3136,512]
  const float* q_w    = (const float*)d_in[4];   // [512,512]
  const float* kv_w   = (const float*)d_in[5];   // [512,1024]
  const float* sr_w   = (const float*)d_in[6];   // [512,512,2,2] OIHW
  const float* sr_b   = (const float*)d_in[7];
  const float* ln_g   = (const float*)d_in[8];
  const float* ln_b   = (const float*)d_in[9];
  const float* proj_w = (const float*)d_in[10];  // [512,512]
  const float* proj_b = (const float*)d_in[11];
  float* out = (float*)d_out;

  char* ws = (char*)d_ws;
  size_t off = 0;
  auto take = [&](size_t bytes) {
    void* p = ws + off;
    off = (off + bytes + 255) & ~(size_t)255;
    return p;
  };
  bf16*  Qb  = (bf16*) take((size_t)12544 * 512 * sizeof(bf16));   // q proj (bf16)
  float* KVp = (float*)take((size_t)3136 * 512 * sizeof(float));   // conv out
  float* KVn = (float*)take((size_t)3136 * 512 * sizeof(float));   // after LN
  bf16*  KVb = (bf16*) take((size_t)3136 * 1024 * sizeof(bf16));   // k|v (bf16)
  float* Of  = (float*)take((size_t)12544 * 512 * sizeof(float));  // attn out

  // 1) q = x @ q_w                 [12544,512] x [512,512] -> bf16
  gemm_wmma_kernel<0, 0, 1, 0><<<dim3(98, 4), 256, 0, stream>>>(
      x, q_w, nullptr, Qb, 12544, 512, 512);

  // 2) SR conv as im2col GEMM     [3136,2048] x [2048,512] -> f32
  gemm_wmma_kernel<1, 1, 0, 0><<<dim3(25, 4), 256, 0, stream>>>(
      v, sr_w, nullptr, KVp, 3136, 512, 2048);

  // 3) LayerNorm (+conv bias)
  ln_kernel<<<3136, 256, 0, stream>>>(KVp, sr_b, ln_g, ln_b, KVn);

  // 4) kv = KVn @ kv_w            [3136,512] x [512,1024] -> bf16
  gemm_wmma_kernel<0, 0, 1, 0><<<dim3(25, 8), 256, 0, stream>>>(
      KVn, kv_w, nullptr, KVb, 3136, 1024, 512);

  // 5) flash attention per (b,h)  (TDM-staged Q/K tiles)
  attention_kernel<<<dim3(49, 32), 128, 0, stream>>>(Qb, KVb, Of);

  // 6) out = O @ proj_w + proj_b  [12544,512] x [512,512] -> f32
  gemm_wmma_kernel<0, 0, 0, 1><<<dim3(98, 4), 256, 0, stream>>>(
      Of, proj_w, proj_b, out, 12544, 512, 512);
}
